// PointerNetwork_47416438948127
// MI455X (gfx1250) — matrix operand order repaired and checked by reading, not compiled, not run on hardware
//
#include <hip/hip_runtime.h>
#include <hip/hip_bf16.h>

// ---------------------------------------------------------------------------
// Pointer network on MI455X (gfx1250, wave32, WMMA).
// Persistent encoder/decoder kernels: LSTM recurrence runs inside one kernel
// with an L2-atomic grid barrier per step; gate weights are pinned in LDS
// (bank-padded, fed to v_wmma_f32_16x16x32_f16 via ds_load_b128); cell state
// stays in registers for the entire recurrence.  6 launches total.
// ---------------------------------------------------------------------------

#define Vv  10000
#define E   256
#define H   512
#define Wd  256
#define S   512
#define T   128
#define B   64
#define G4H 2048          // 4*H
#define KCAT 768          // E + H (encoder fused K)
#define BH  (B * H)       // 32768

#define ENC_WSTRIDE 776   // 768 + 8 pad halves  (stride % 64 banks == 4 dwords)
#define DEC_WSTRIDE 520   // 512 + 8 pad halves  (stride % 64 banks == 4 dwords)

typedef __attribute__((ext_vector_type(16))) _Float16 v16h;
typedef __attribute__((ext_vector_type(8)))  float    v8f;

// ---------------- transcendentals ------------------------------------------
__device__ __forceinline__ float fast_tanh(float x) {
#if __has_builtin(__builtin_amdgcn_tanhf)
  return __builtin_amdgcn_tanhf(x);
#elif __has_builtin(__builtin_amdgcn_tanh_f32)
  return __builtin_amdgcn_tanh_f32(x);
#else
  return tanhf(x);
#endif
}
__device__ __forceinline__ float fast_sigmoid(float x) {
  return __builtin_fmaf(0.5f, fast_tanh(0.5f * x), 0.5f);
}

// ---------------- WMMA fragment loaders (contiguous k order: k = 16*(lane/16) + s) ----------------
__device__ __forceinline__ v16h load_a_frag(const _Float16* rowk, int half8) {
  v16h a;
  float4* ap = reinterpret_cast<float4*>(&a);
  ap[0] = *reinterpret_cast<const float4*>(rowk + 2 * half8);
  ap[1] = *reinterpret_cast<const float4*>(rowk + 2 * half8 + 8);
  return a;
}
__device__ __forceinline__ v16h load_b_frag(const _Float16* p) {
  v16h b;
  float4* bp = reinterpret_cast<float4*>(&b);
  const float4* q = reinterpret_cast<const float4*>(p);
  bp[0] = q[0];
  bp[1] = q[1];
  return b;
}

// ---------------- device-wide barrier (monotonic counter in L2) -------------
__device__ __forceinline__ void grid_barrier(unsigned* bar, unsigned target) {
  __syncthreads();
  __threadfence();                       // release: flush this WGP's stores
  if (threadIdx.x == 0) {
    __hip_atomic_fetch_add(bar, 1u, __ATOMIC_RELEASE, __HIP_MEMORY_SCOPE_AGENT);
    while (__hip_atomic_load(bar, __ATOMIC_ACQUIRE, __HIP_MEMORY_SCOPE_AGENT) <
           target)
      __builtin_amdgcn_s_sleep(4);
  }
  __syncthreads();
  __threadfence();                       // acquire: invalidate stale lines
}

// ---------------- prep kernels ---------------------------------------------
__global__ __launch_bounds__(256) void k_embed(const int* __restrict__ tokens,
                                               const float* __restrict__ table,
                                               _Float16* __restrict__ emb16) {
  int m = blockIdx.x;             // flat row = s*B + b  (time-major)
  int s = m >> 6, b = m & 63;
  int tok = tokens[b * S + s];
  int e = threadIdx.x;
  emb16[(size_t)m * E + e] = (_Float16)table[(size_t)tok * E + e];
}

__global__ __launch_bounds__(256) void k_prep(
    const float* __restrict__ eWih, const float* __restrict__ eWhh,
    const float* __restrict__ ebih, const float* __restrict__ ebhh,
    const float* __restrict__ dWhh, const float* __restrict__ dbih,
    const float* __restrict__ dbhh, const float* __restrict__ W1w,
    _Float16* __restrict__ Wcat, _Float16* __restrict__ Wdec,
    _Float16* __restrict__ W1h, float* __restrict__ bias_e,
    float* __restrict__ bias_d) {
  int n = blockIdx.x;       // 0..2047 (gate-row)
  int t = threadIdx.x;      // 0..255
  Wcat[(size_t)n * KCAT + t]       = (_Float16)eWih[(size_t)n * E + t];
  Wcat[(size_t)n * KCAT + 256 + t] = (_Float16)eWhh[(size_t)n * H + t];
  Wcat[(size_t)n * KCAT + 512 + t] = (_Float16)eWhh[(size_t)n * H + 256 + t];
  Wdec[(size_t)n * H + t]          = (_Float16)dWhh[(size_t)n * H + t];
  Wdec[(size_t)n * H + 256 + t]    = (_Float16)dWhh[(size_t)n * H + 256 + t];
  if (n < Wd) {
    W1h[(size_t)n * H + t]       = (_Float16)W1w[(size_t)n * H + t];
    W1h[(size_t)n * H + 256 + t] = (_Float16)W1w[(size_t)n * H + 256 + t];
  }
  if (t == 0) {
    bias_e[n] = ebih[n] + ebhh[n];
    bias_d[n] = dbih[n] + dbhh[n];
  }
}

__global__ __launch_bounds__(256) void k_init(_Float16* __restrict__ h16_0,
                                              unsigned* __restrict__ bars) {
  int i = blockIdx.x * 256 + threadIdx.x;   // BH
  h16_0[i] = (_Float16)0.0f;
  if (i < 64) bars[i] = 0u;   // zero the whole counter block on every replay
}

// ---------------- persistent encoder ----------------------------------------
// 32 WGs x 128 threads; WG owns one 16-col j-tile of H for all 4 gates.
// Gate weights pinned in LDS; c state in registers; 512 steps, 1 barrier/step.
__global__ __launch_bounds__(128) void k_enc_persist(
    const _Float16* __restrict__ emb16, const _Float16* __restrict__ Wcat,
    const float* __restrict__ bias, _Float16* __restrict__ h16,
    float* __restrict__ h32_final, _Float16* __restrict__ encS16,
    unsigned* bar) {
  __shared__ _Float16 wlds[64 * ENC_WSTRIDE];   // 99,328 B

  const int tid = threadIdx.x;
  const int j0 = blockIdx.x * 16;
  // stage this WG's weight slice: rows r = g*16 + jl  ->  Wcat[g*H + j0 + jl]
  for (int idx = tid; idx < 64 * 96; idx += 128) {
    int r = idx / 96, c = idx % 96;
    int grow = (r >> 4) * H + j0 + (r & 15);
    *reinterpret_cast<float4*>(&wlds[r * ENC_WSTRIDE + c * 8]) =
        *reinterpret_cast<const float4*>(Wcat + (size_t)grow * KCAT +
                                         (size_t)c * 8);
  }
  __syncthreads();

  const int lane = tid & 31, wave = tid >> 5;
  const int nlane = lane & 15;
  const int half8 = (lane >> 4) * 8, half16 = (lane >> 4) * 16;
  const int m0 = wave * 16;         // batch row tile
  const int ar = m0 + nlane;        // this lane's A row
  const int col = j0 + nlane;
  const float bi  = bias[0 * H + col];
  const float bff = bias[1 * H + col];
  const float bg  = bias[2 * H + col];
  const float bo  = bias[3 * H + col];

  float creg[8];
#pragma unroll
  for (int r = 0; r < 8; ++r) creg[r] = 0.0f;

  unsigned epoch = 0;
  for (int s = 0; s < S; ++s) {
    const _Float16* hin = h16 + (s & 1) * BH;
    _Float16* hout = h16 + ((s + 1) & 1) * BH;

    v8f zero = {};
    v8f acc[4] = {zero, zero, zero, zero};
#pragma unroll
    for (int ks = 0; ks < KCAT / 32; ++ks) {
      const int kbase = ks * 32;
      const _Float16* rowk =
          (kbase < E) ? emb16 + ((size_t)s * B + ar) * E + kbase
                      : hin + (size_t)ar * H + (kbase - E);
      v16h a = load_a_frag(rowk, half8);
      // separate B-fragment registers -> ds_loads clause + overlap with WMMA
      v16h bfr[4];
#pragma unroll
      for (int g = 0; g < 4; ++g)
        bfr[g] = load_b_frag(&wlds[(g * 16 + nlane) * ENC_WSTRIDE + kbase + half16]);
#pragma unroll
      for (int g = 0; g < 4; ++g)
        acc[g] = __builtin_amdgcn_wmma_f32_16x16x32_f16(
            false, a, false, bfr[g], (short)0, acc[g], false, false);
    }
#pragma unroll
    for (int r = 0; r < 8; ++r) {
      const int brow = m0 + ((lane < 16) ? r : (r + 8));
      const size_t idx = (size_t)brow * H + col;
      const float gi = fast_sigmoid(acc[0][r] + bi);
      const float gf = fast_sigmoid(acc[1][r] + bff);
      const float gg = fast_tanh(acc[2][r] + bg);
      const float go = fast_sigmoid(acc[3][r] + bo);
      const float c = gf * creg[r] + gi * gg;
      const float h = go * fast_tanh(c);
      creg[r] = c;
      const _Float16 hh = (_Float16)h;
      hout[idx] = hh;
      encS16[((size_t)s * B + brow) * H + col] = hh;
      if (s == S - 1) h32_final[idx] = h;   // c0 for the decoder
    }
    grid_barrier(bar, (++epoch) * (unsigned)gridDim.x);
  }
}

// ---------------- blend1 = encS @ W1^T + b1  (one-shot WMMA GEMM) -----------
__global__ __launch_bounds__(128) void k_blend1(const _Float16* __restrict__ encS,
                                                const _Float16* __restrict__ W1h,
                                                const float* __restrict__ W1b,
                                                float* __restrict__ blend1) {
  const int lane   = threadIdx.x & 31;
  const int wave   = threadIdx.x >> 5;
  const int nlane  = lane & 15;
  const int half8  = (lane >> 4) * 8;
  const int half16 = (lane >> 4) * 16;
  const int m0 = blockIdx.x * 16;                      // flat row tile
  const _Float16* arow = encS + (size_t)(m0 + nlane) * H;
  for (int nt = wave; nt < 16; nt += 4) {
    const int n0 = nt * 16;
    v8f acc = {};
    const _Float16* brow = W1h + (size_t)(n0 + nlane) * H + half16;
#pragma unroll
    for (int ks = 0; ks < H / 32; ++ks) {
      const int kbase = ks * 32;
      v16h a  = load_a_frag(arow + kbase, half8);
      v16h bf = load_b_frag(brow + kbase);
      acc = __builtin_amdgcn_wmma_f32_16x16x32_f16(
          false, a, false, bf, (short)0, acc, false, false);
    }
    const float bias = W1b[n0 + nlane];
#pragma unroll
    for (int r = 0; r < 8; ++r) {
      const int m = m0 + ((lane < 16) ? r : (r + 8));
      blend1[(size_t)m * Wd + n0 + nlane] = acc[r] + bias;
    }
  }
}

// ---------------- persistent decoder ----------------------------------------
// 64 WGs x 256 threads.  WGs 0..31 run the WMMA LSTM cell (weights in LDS,
// c in registers); after one grid barrier all 64 WGs run attention +
// log-softmax for their batch row.  128 steps, 1 barrier/step.
__global__ __launch_bounds__(256) void k_dec_persist(
    const _Float16* __restrict__ Wdec, const float* __restrict__ bias,
    const float* __restrict__ dec_h0, const float* __restrict__ h32_encfinal,
    _Float16* __restrict__ h16, float* __restrict__ h32,
    const float* __restrict__ blend1, const float* __restrict__ W2w,
    const float* __restrict__ W2b, const float* __restrict__ vtw,
    const float* __restrict__ vtb, float* __restrict__ out, unsigned* bar) {
  __shared__ _Float16 wlds[64 * DEC_WSTRIDE];   // 66,560 B
  __shared__ float sh_h[H];
  __shared__ float sh_b2[Wd];
  __shared__ float sh_vt[Wd];
  __shared__ float sh_u[S];
  __shared__ float red[256];

  const int tid = threadIdx.x;
  const int wg = blockIdx.x;
  const bool cellwg = (wg < 32);

  // stage cell weights (WGs 0..31): rows r = g*16 + jl -> Wdec[g*H + j0 + jl]
  if (cellwg) {
    const int j0 = wg * 16;
    for (int idx = tid; idx < 64 * 64; idx += 256) {
      int r = idx >> 6, c = idx & 63;
      int grow = (r >> 4) * H + j0 + (r & 15);
      *reinterpret_cast<float4*>(&wlds[r * DEC_WSTRIDE + c * 8]) =
          *reinterpret_cast<const float4*>(Wdec + (size_t)grow * H +
                                           (size_t)c * 8);
    }
  }
  // init decoder h buffers (all WGs cooperate)
  for (int i = wg * 256 + tid; i < BH; i += 64 * 256) {
    float hv = dec_h0[i];
    h32[i] = hv;
    h16[i] = (_Float16)hv;
  }

  const int lane = tid & 31, wave = tid >> 5;
  const int nlane = lane & 15;
  const int half8 = (lane >> 4) * 8, half16 = (lane >> 4) * 16;
  const int m0 = (wave & 3) * 16;
  const int colj = (wg & 31) * 16 + nlane;
  const bool cellthr = cellwg && (wave < 4);

  float creg[8];
  float bi = 0.f, bff = 0.f, bg = 0.f, bo = 0.f;
#pragma unroll
  for (int r = 0; r < 8; ++r) creg[r] = 0.0f;
  if (cellthr) {
    bi  = bias[0 * H + colj];
    bff = bias[1 * H + colj];
    bg  = bias[2 * H + colj];
    bo  = bias[3 * H + colj];
#pragma unroll
    for (int r = 0; r < 8; ++r) {
      const int brow = m0 + ((lane < 16) ? r : (r + 8));
      creg[r] = h32_encfinal[(size_t)brow * H + colj];   // c0 = last enc h
    }
  }

  unsigned epoch = 0;
  grid_barrier(bar, (++epoch) * (unsigned)gridDim.x);    // weights + h init

  const float vb = vtb[0];
  for (int t = 0; t < T; ++t) {
    const _Float16* hin = h16 + (t & 1) * BH;
    _Float16* hout = h16 + ((t + 1) & 1) * BH;
    float* h32out = h32 + ((t + 1) & 1) * BH;

    // ---- LSTM cell phase (x input is identically zero) ----
    if (cellthr) {
      v8f zero = {};
      v8f acc[4] = {zero, zero, zero, zero};
      const int ar = m0 + nlane;
#pragma unroll
      for (int ks = 0; ks < H / 32; ++ks) {
        const int kbase = ks * 32;
        v16h a = load_a_frag(hin + (size_t)ar * H + kbase, half8);
        v16h bfr[4];
#pragma unroll
        for (int g = 0; g < 4; ++g)
          bfr[g] = load_b_frag(&wlds[(g * 16 + nlane) * DEC_WSTRIDE + kbase + half16]);
#pragma unroll
        for (int g = 0; g < 4; ++g)
          acc[g] = __builtin_amdgcn_wmma_f32_16x16x32_f16(
              false, a, false, bfr[g], (short)0, acc[g], false, false);
      }
#pragma unroll
      for (int r = 0; r < 8; ++r) {
        const int brow = m0 + ((lane < 16) ? r : (r + 8));
        const size_t idx = (size_t)brow * H + colj;
        const float gi = fast_sigmoid(acc[0][r] + bi);
        const float gf = fast_sigmoid(acc[1][r] + bff);
        const float gg = fast_tanh(acc[2][r] + bg);
        const float go = fast_sigmoid(acc[3][r] + bo);
        const float c = gf * creg[r] + gi * gg;
        const float h = go * fast_tanh(c);
        creg[r] = c;
        hout[idx] = (_Float16)h;
        h32out[idx] = h;
      }
    }
    grid_barrier(bar, (++epoch) * (unsigned)gridDim.x);

    // ---- attention + log-softmax for batch row b = wg ----
    const int b = wg;
    for (int i = tid; i < H; i += 256) sh_h[i] = h32out[(size_t)b * H + i];
    __syncthreads();
    {
      float acc = W2b[tid];
      const float* wrow = W2w + (size_t)tid * H;
      for (int k = 0; k < H; ++k) acc = __builtin_fmaf(sh_h[k], wrow[k], acc);
      sh_b2[tid] = acc;
      sh_vt[tid] = vtw[tid];
    }
    __syncthreads();

    const int wv = tid >> 5, ln = tid & 31;
    for (int s0 = wv; s0 < S; s0 += 8) {
      const float* brow = blend1 + ((size_t)s0 * B + b) * Wd;
      float sum = 0.0f;
#pragma unroll
      for (int i = 0; i < 8; ++i) {
        const int w = ln + i * 32;
        sum = __builtin_fmaf(fast_tanh(brow[w] + sh_b2[w]), sh_vt[w], sum);
      }
      for (int off = 16; off; off >>= 1) sum += __shfl_down(sum, off, 32);
      if (ln == 0) sh_u[s0] = sum + vb;
    }
    __syncthreads();

    float m = -1e30f;
    for (int s = tid; s < S; s += 256) m = fmaxf(m, sh_u[s]);
    red[tid] = m;
    __syncthreads();
    for (int off = 128; off; off >>= 1) {
      if (tid < off) red[tid] = fmaxf(red[tid], red[tid + off]);
      __syncthreads();
    }
    const float mx = red[0];
    __syncthreads();
    float e = 0.0f;
    for (int s = tid; s < S; s += 256) e += __expf(sh_u[s] - mx);
    red[tid] = e;
    __syncthreads();
    for (int off = 128; off; off >>= 1) {
      if (tid < off) red[tid] += red[tid + off];
      __syncthreads();
    }
    const float lse = __logf(red[0]) + mx;
    float* orow = out + ((size_t)b * T + t) * S;
    for (int s = tid; s < S; s += 256) orow[s] = sh_u[s] - lse;
    __syncthreads();   // protect sh_u/red before next iteration
  }
}

// ---------------------------------------------------------------------------
extern "C" void kernel_launch(void* const* d_in, const int* in_sizes, int n_in,
                              void* d_out, int out_size, void* d_ws, size_t ws_size,
                              hipStream_t stream) {
  (void)in_sizes; (void)n_in; (void)out_size; (void)ws_size;
  const int*   tokens = (const int*)  d_in[0];
  const float* dec_h0 = (const float*)d_in[1];
  const float* table  = (const float*)d_in[2];
  const float* eWih   = (const float*)d_in[3];
  const float* eWhh   = (const float*)d_in[4];
  const float* ebih   = (const float*)d_in[5];
  const float* ebhh   = (const float*)d_in[6];
  /* d_in[7] = dec_Wih: unused — decoder input is identically zero */
  const float* dWhh   = (const float*)d_in[8];
  const float* dbih   = (const float*)d_in[9];
  const float* dbhh   = (const float*)d_in[10];
  const float* W1w    = (const float*)d_in[11];
  const float* W1b    = (const float*)d_in[12];
  const float* W2w    = (const float*)d_in[13];
  const float* W2b    = (const float*)d_in[14];
  const float* vtw    = (const float*)d_in[15];
  const float* vtb    = (const float*)d_in[16];
  float* out = (float*)d_out;

  char* wsb = (char*)d_ws;
  size_t off = 0;
  auto alloc = [&](size_t bytes) -> void* {
    void* p = wsb + off;
    off = (off + bytes + 255) & ~(size_t)255;
    return p;
  };
  _Float16* emb16  = (_Float16*)alloc((size_t)S * B * E * 2);      // 16.8 MB
  _Float16* Wcat   = (_Float16*)alloc((size_t)G4H * KCAT * 2);     //  3.1 MB
  _Float16* Wdec   = (_Float16*)alloc((size_t)G4H * H * 2);        //  2.1 MB
  _Float16* W1h    = (_Float16*)alloc((size_t)Wd * H * 2);         //  0.3 MB
  float*    bias_e = (float*)   alloc((size_t)G4H * 4);
  float*    bias_d = (float*)   alloc((size_t)G4H * 4);
  _Float16* h16    = (_Float16*)alloc((size_t)2 * BH * 2);         // double buf
  float*    h32    = (float*)   alloc((size_t)2 * BH * 4);         // dec double buf
  float*    h32fin = (float*)   alloc((size_t)BH * 4);             // enc final h
  _Float16* encS16 = (_Float16*)alloc((size_t)S * B * H * 2);      // 33.6 MB
  float*    blend1 = (float*)   alloc((size_t)S * B * Wd * 4);     // 33.6 MB
  unsigned* bars   = (unsigned*)alloc(256);  // 64 uints; [0]=enc, [32]=dec

  // 1) embedding gather + f16 cast (time-major [S*B, E])
  k_embed<<<S * B, 256, 0, stream>>>(tokens, table, emb16);
  // 2) weight packing ([N][K] f16 rows for WMMA-B) + fused biases
  k_prep<<<G4H, 256, 0, stream>>>(eWih, eWhh, ebih, ebhh, dWhh, dbih, dbhh, W1w,
                                  Wcat, Wdec, W1h, bias_e, bias_d);
  // 3) h0 = 0, zero the barrier counter block (re-zeroed on every replay)
  k_init<<<BH / 256, 256, 0, stream>>>(h16, bars);

  // 4) persistent encoder: 512 steps, grid barrier per step
  k_enc_persist<<<H / 16, 128, 0, stream>>>(emb16, Wcat, bias_e, h16, h32fin,
                                            encS16, bars + 0);

  // 5) blend1 = encS @ W1^T + b1  (hoisted out of decoder loop)
  k_blend1<<<(S * B) / 16, 128, 0, stream>>>(encS16, W1h, W1b, blend1);

  // 6) persistent decoder: 128 steps of (WMMA cell -> attention/log-softmax)
  k_dec_persist<<<B, 256, 0, stream>>>(Wdec, bias_d, dec_h0, h32fin, h16, h32,
                                       blend1, W2w, W2b, vtw, vtb, out,
                                       bars + 32);
}
